// BigramModel_37606733643795
// MI455X (gfx1250) — matrix operand order repaired and checked by reading, not compiled, not run on hardware
//
#include <hip/hip_runtime.h>
#include <hip/hip_bf16.h>

// ---- model dims (match reference) ----
constexpr int Bb = 128, T = 256, Vv_ = 100, C = 384, H = 6, HD = 64, L = 6;
constexpr int FF = 4 * C;            // 1536
constexpr int M = Bb * T;            // 32768 rows in the flattened [B*T, C] stream

typedef __bf16 bf16;
typedef __attribute__((ext_vector_type(16))) __bf16 v16bf;
typedef __attribute__((ext_vector_type(8)))  float  v8f;
typedef __attribute__((ext_vector_type(4)))  int    v4i;

// ---- CDNA5 data-movement feature detection (device pass only) ----
#if defined(__AMDGCN__) && __has_builtin(__builtin_amdgcn_global_load_async_to_lds_b128) && \
    __has_builtin(__builtin_amdgcn_s_wait_asynccnt)
#define HAVE_ASYNC 1
#endif
#if defined(__AMDGCN__) && __has_builtin(__builtin_amdgcn_tensor_load_to_lds) && \
    __has_builtin(__builtin_amdgcn_s_wait_tensorcnt)
#define HAVE_TDM 1
#endif

// ---------------------------------------------------------------------------
// WMMA helpers (CDNA5 wave32, D = A(16x32 bf16) * B(32x16 bf16) + C(16x16 f32))
// ---------------------------------------------------------------------------
__device__ inline v8f wmma_bf16(v16bf a, v16bf b, v8f c) {
  return __builtin_amdgcn_wmma_f32_16x16x32_bf16(
      /*neg_a=*/false, a, /*neg_b=*/false, b,
      /*c_mod=*/(short)0, c, /*reuse_a=*/false, /*reuse_b=*/false);
}

// Load a 32-byte fragment as two 16-byte chunks (ds_load_b128 x2 from LDS).
__device__ inline v16bf ldfrag(const bf16* p0, const bf16* p1) {
  union { uint4 u[2]; v16bf v; } r;
  r.u[0] = *reinterpret_cast<const uint4*>(p0);
  r.u[1] = *reinterpret_cast<const uint4*>(p1);
  return r.v;
}

// A-matrix 16x32 bf16 layout: lanes 0-15 row M=lane hold K 0..15;
// lanes 16-31 hold K 16..31 (contiguous, matching B).
__device__ inline v16bf ldfragA(const bf16* rowbase, int akb) {
  return ldfrag(rowbase + 2 * akb, rowbase + 2 * akb + 8);
}
// B-matrix 32x16 bf16: lane n (0-15) = column n holds K 0..15; lane n+16 holds
// K 16..31 (contiguous per lane when stored K-major per column).
__device__ inline v16bf ldfragB(const bf16* colbase, int bkb) {
  return ldfrag(colbase + bkb, colbase + bkb + 8);
}

// ---------------------------------------------------------------------------
// CDNA5 async / TDM copies
// ---------------------------------------------------------------------------
// 16B straight global->LDS copy, one per lane (GLOBAL_LOAD_ASYNC_TO_LDS_B128).
// Builtin signature (from compiler diagnostic): (v4i AS1*, v4i AS3*, imm, imm).
__device__ inline void async_copy_b128(void* lds_dst, const void* gsrc) {
#ifdef HAVE_ASYNC
  __builtin_amdgcn_global_load_async_to_lds_b128(
      (__attribute__((address_space(1))) v4i*)gsrc,
      (__attribute__((address_space(3))) v4i*)lds_dst, 0, 0);
#else
  *reinterpret_cast<uint4*>(lds_dst) = *reinterpret_cast<const uint4*>(gsrc);
#endif
}
__device__ inline void async_join() {
#ifdef HAVE_ASYNC
  __builtin_amdgcn_s_wait_asynccnt(0);
#endif
}

// Tensor Data Mover: DMA a [rows x rowElems] bf16 tile (global row stride
// strideElems) into LDS, inserting `padAmount+1` DWORDs of padding every
// 2^(padInterval+1) DWORDs (-> padded LDS row stride).  2D descriptor per
// CDNA5 ISA.
__device__ inline void tdm_load_tile_bf16(void* lds_dst, const void* gsrc,
                                          int rowElems, int rows, int strideElems,
                                          int padInterval, int padAmount) {
#ifdef HAVE_TDM
  typedef unsigned int u32x4 __attribute__((ext_vector_type(4)));
  typedef int i32x4 __attribute__((ext_vector_type(4)));
  typedef int i32x8 __attribute__((ext_vector_type(8)));
  unsigned lds_off = (unsigned)(unsigned long long)
      (__attribute__((address_space(3))) void*)lds_dst;
  unsigned long long ga = (unsigned long long)gsrc;
  u32x4 g0;
  g0[0] = 1u;                                        // count=1 (user descriptor)
  g0[1] = lds_off;                                   // lds_addr
  g0[2] = (unsigned)ga;                              // global_addr[31:0]
  g0[3] = (unsigned)((ga >> 32) & 0x01ffffffu)       // global_addr[56:32]
        | 0x80000000u;                               // type=2 ("image")
  i32x8 g1;
  g1[0] = (1 << 16)                                  // data_size = 2 bytes
        | ((padAmount | padInterval) ? (1 << 20) : 0)// pad_enable
        | (padInterval << 22) | (padAmount << 25);
  g1[1] = (strideElems & 0xffff) << 16;              // tensor_dim0[15:0]
  g1[2] = ((strideElems >> 16) & 0xffff)             // tensor_dim0[31:16]
        | ((rows & 0xffff) << 16);                   // tensor_dim1[15:0]
  g1[3] = ((rows >> 16) & 0xffff)                    // tensor_dim1[31:16]
        | ((rowElems & 0xffff) << 16);               // tile_dim0
  g1[4] = rows & 0xffff;                             // tile_dim1 (tile_dim2 = 0)
  g1[5] = strideElems;                               // tensor_dim0_stride[31:0]
  g1[6] = 0;                                         // stride hi / dim1_stride lo
  g1[7] = 0;
  i32x4 z4 = {0, 0, 0, 0};
#if __has_include(<hip/amd_detail/amd_gfx1250_TDM.h>)
  i32x8 z8 = {0, 0, 0, 0, 0, 0, 0, 0};
  __builtin_amdgcn_tensor_load_to_lds(g0, g1, z4, z4, z8, 0);  // clang-23: 6 args
#else
  __builtin_amdgcn_tensor_load_to_lds(g0, g1, z4, z4, 0);      // ROCm 7.2: 5 args
#endif
#else
  (void)lds_dst; (void)gsrc; (void)rowElems; (void)rows;
  (void)strideElems; (void)padInterval; (void)padAmount;
#endif
}
__device__ inline void tdm_join() {
#ifdef HAVE_TDM
  __builtin_amdgcn_s_wait_tensorcnt(0);
#endif
}

// ---------------------------------------------------------------------------
// fp32 -> bf16 conversion (weights)
// ---------------------------------------------------------------------------
__global__ void k_f32_to_bf16(const float* __restrict__ in, bf16* __restrict__ out, size_t n) {
  size_t i = (size_t)blockIdx.x * blockDim.x + threadIdx.x;
  size_t stride = (size_t)gridDim.x * blockDim.x;
  for (; i < n; i += stride) out[i] = (bf16)in[i];
}

// ---------------------------------------------------------------------------
// Embedding: x[b,t,:] = embed[idx[b,t],:] + pos[t,:]
// ---------------------------------------------------------------------------
__global__ void k_embed(const int* __restrict__ idx, const float* __restrict__ emb,
                        const float* __restrict__ pos, float* __restrict__ x) {
  int m = blockIdx.x;
  int t = m % T;
  int id = idx[m];
  for (int c = threadIdx.x; c < C; c += blockDim.x)
    x[(size_t)m * C + c] = emb[(size_t)id * C + c] + pos[(size_t)t * C + c];
}

// ---------------------------------------------------------------------------
// LayerNorm: one wave per row, pure shuffle reduction (wave32).  fp32 in,
// bf16 out (GEMM operand precision).
// ---------------------------------------------------------------------------
__global__ __launch_bounds__(256) void k_layernorm(
    const float* __restrict__ x, const float* __restrict__ g,
    const float* __restrict__ b, bf16* __restrict__ out) {
  int row  = blockIdx.x * 8 + (threadIdx.x >> 5);
  int lane = threadIdx.x & 31;
  const float* xr = x + (size_t)row * C;
  float v[C / 32];
  float s = 0.f;
#pragma unroll
  for (int i = 0; i < C / 32; ++i) { v[i] = xr[lane + 32 * i]; s += v[i]; }
#pragma unroll
  for (int mk = 1; mk < 32; mk <<= 1) s += __shfl_xor(s, mk, 32);
  float mu = s * (1.0f / C);
  float var = 0.f;
#pragma unroll
  for (int i = 0; i < C / 32; ++i) { float d = v[i] - mu; var += d * d; }
#pragma unroll
  for (int mk = 1; mk < 32; mk <<= 1) var += __shfl_xor(var, mk, 32);
  float rstd = rsqrtf(var * (1.0f / C) + 1e-5f);
  bf16* orow = out + (size_t)row * C;
#pragma unroll
  for (int i = 0; i < C / 32; ++i) {
    int c = lane + 32 * i;
    orow[c] = (bf16)((v[i] - mu) * rstd * g[c] + b[c]);
  }
}

// ---------------------------------------------------------------------------
// Tiled bf16 GEMM with fused epilogues and double-buffered LDS staging.
//   D[M,N] = epi(A[M,K] @ W[K,N] + bias)
// EPI: 0 = bf16 out (QKV)             1 = bf16 out, +bias, ReLU (MLP1)
//      2 = f32 residual  Df += (+bias) (out-proj / MLP2)
//      3 = f32 out, +bias (LM head, N edge-guarded)
// Block 256 threads (8 waves); block tile 128x64; K staged 64/iter.  Tile i+1
// is DMA'd (TDM / async-to-LDS) into the ping-pong buffer while tile i feeds
// the WMMA pipe; one s_wait_asynccnt/tensorcnt + barrier per K step.
// ---------------------------------------------------------------------------
template <int EPI>
__global__ __launch_bounds__(256) void k_gemm_bf16(
    const bf16* __restrict__ A, const bf16* __restrict__ W,
    const float* __restrict__ bias, bf16* __restrict__ Dbf,
    float* __restrict__ Df, int N, int K) {
  constexpr int BM = 128, BN = 64, BK = 64, PAD = 8;
  __shared__ __align__(16) bf16 As[2][BM][BK + PAD];
  __shared__ __align__(16) bf16 Bs[2][BN][BK + PAD];  // [n][k] (K-major per column)

  const int tid = threadIdx.x, wave = tid >> 5, lane = tid & 31;
  const int wm = wave & 3, wn = wave >> 2;          // 4 waves along M, 2 along N
  const int m0 = blockIdx.y * BM, n0 = blockIdx.x * BN;
  const int arow = lane & 15, akb = (lane < 16) ? 0 : 8;
  const int bkb  = (lane < 16) ? 0 : 16;
  const int halfofs = (lane >= 16) ? 8 : 0;

  v8f acc[2][2] = {};

  auto stageA = [&](int buf, int k0) {
#ifdef HAVE_TDM
    if (wave == 0) {
      // pad_interval=4 -> pad every 2^(4+1)=32 DWORDs (=64 bf16 row);
      // pad_amount=3 -> 4 DWORDs (=8 bf16) padding => LDS row stride 72.
      tdm_load_tile_bf16(&As[buf][0][0], &A[(size_t)m0 * K + k0], BK, BM, K, 4, 3);
    }
#else
    for (int c = tid; c < BM * (BK / 8); c += 256) {
      int r = c >> 3, cc = (c & 7) * 8;
      async_copy_b128(&As[buf][r][cc], &A[(size_t)(m0 + r) * K + k0 + cc]);
    }
#endif
  };
  auto stageB = [&](int buf, int k0) {
    for (int c = tid; c < BK * (BN / 8); c += 256) {
      int kk = c >> 3, nn = (c & 7) * 8;
      int gn = n0 + nn;
      union { uint4 u; bf16 e[8]; } vr;
      if (((N & 7) == 0) && (gn + 7 < N)) {
        vr.u = *reinterpret_cast<const uint4*>(&W[(size_t)(k0 + kk) * N + gn]);
      } else {
#pragma unroll
        for (int e = 0; e < 8; ++e)
          vr.e[e] = (gn + e < N) ? W[(size_t)(k0 + kk) * N + gn + e] : (bf16)0.0f;
      }
#pragma unroll
      for (int e = 0; e < 8; ++e) Bs[buf][nn + e][kk] = vr.e[e];
    }
  };
  auto join_stages = [&]() {
#ifdef HAVE_TDM
    if (wave == 0) tdm_join();
#endif
    async_join();
    __syncthreads();
  };

  stageA(0, 0);
  stageB(0, 0);
  join_stages();

  const int nk = K / BK;
  for (int i = 0; i < nk; ++i) {
    const int cur = i & 1;
    if (i + 1 < nk) {             // overlap next tile's DMA with this tile's WMMAs
      stageA(cur ^ 1, (i + 1) * BK);
      stageB(cur ^ 1, (i + 1) * BK);
    }
#pragma unroll
    for (int ks = 0; ks < BK; ks += 32) {
      v16bf af[2], bf_[2];
#pragma unroll
      for (int ii = 0; ii < 2; ++ii)
        af[ii] = ldfragA(&As[cur][wm * 32 + ii * 16 + arow][ks], akb);
#pragma unroll
      for (int j = 0; j < 2; ++j)
        bf_[j] = ldfragB(&Bs[cur][wn * 32 + j * 16 + (lane & 15)][ks], bkb);
#pragma unroll
      for (int ii = 0; ii < 2; ++ii)
#pragma unroll
        for (int j = 0; j < 2; ++j)
          acc[ii][j] = wmma_bf16(af[ii], bf_[j], acc[ii][j]);
    }
    join_stages();
  }

  // --- epilogue (C/D layout: elem e -> row = base+e+(lane>=16?8:0), col = lane&15) ---
#pragma unroll
  for (int i = 0; i < 2; ++i) {
#pragma unroll
    for (int j = 0; j < 2; ++j) {
      int col = n0 + wn * 32 + j * 16 + (lane & 15);
      float bval = 0.f;
      if constexpr (EPI != 0) bval = (col < N) ? bias[col] : 0.f;
#pragma unroll
      for (int e = 0; e < 8; ++e) {
        int row = m0 + wm * 32 + i * 16 + e + halfofs;
        float vvv = acc[i][j][e] + bval;
        if constexpr (EPI == 0) {
          Dbf[(size_t)row * N + col] = (bf16)vvv;
        } else if constexpr (EPI == 1) {
          Dbf[(size_t)row * N + col] = (bf16)fmaxf(vvv, 0.f);
        } else if constexpr (EPI == 2) {
          if (col < N) Df[(size_t)row * N + col] += vvv;
        } else {
          if (col < N) Df[(size_t)row * N + col] = vvv;
        }
      }
    }
  }
}

// ---------------------------------------------------------------------------
// Flash attention: one WG per (64-query tile, head, batch).  4 waves; each
// wave owns 16 query rows.  S = Q K^T and O += P V both via WMMA bf16.
// K/V tiles are double-buffered: block j+1's async K-copy and V transpose
// overlap with block j's WMMAs + online softmax.  Width-16 shuffles match the
// 16-lane halves of the C/D register layout.
// ---------------------------------------------------------------------------
__global__ __launch_bounds__(128) void k_attention(
    const bf16* __restrict__ Q, const bf16* __restrict__ Kt,
    const bf16* __restrict__ Vt, bf16* __restrict__ O) {
  __shared__ __align__(16) bf16 Qs[64][72];
  __shared__ __align__(16) bf16 Ks[2][64][72];     // [key][hd]
  __shared__ __align__(16) bf16 Vs[2][64][72];     // transposed: [hd][key]
  __shared__ __align__(16) bf16 Ps[4][16][72];     // per-wave P scratch

  const int qt = blockIdx.x, hh = blockIdx.y, bb = blockIdx.z;
  const int tid = threadIdx.x, wave = tid >> 5, lane = tid & 31;
  const int arow = lane & 15, akb = (lane < 16) ? 0 : 8;
  const int bkb = (lane < 16) ? 0 : 16;
  const int halfofs = (lane >= 16) ? 8 : 0;
  const size_t base = ((size_t)bb * T) * C + hh * HD;

  auto stageKV = [&](int buf, int j) {
    for (int c = tid; c < 64 * 8; c += 128) {
      int r = c >> 3, cc = (c & 7) * 8;
      async_copy_b128(&Ks[buf][r][cc], &Kt[base + (size_t)(j * 64 + r) * C + cc]);
      union { uint4 u; bf16 e8[8]; } vr;
      vr.u = *reinterpret_cast<const uint4*>(&Vt[base + (size_t)(j * 64 + r) * C + cc]);
#pragma unroll
      for (int ee = 0; ee < 8; ++ee) Vs[buf][cc + ee][r] = vr.e8[ee];  // transpose
    }
  };

  for (int c = tid; c < 64 * 8; c += 128) {
    int r = c >> 3, cc = (c & 7) * 8;
    async_copy_b128(&Qs[r][cc], &Q[base + (size_t)(qt * 64 + r) * C + cc]);
  }
  stageKV(0, 0);
  async_join();
  __syncthreads();

  float mrow[8], lrow[8];
  v8f oacc[4] = {};
#pragma unroll
  for (int e = 0; e < 8; ++e) { mrow[e] = -1e30f; lrow[e] = 0.f; }

  for (int j = 0; j <= qt; ++j) {
    const int cur = j & 1;
    if (j + 1 <= qt) stageKV(cur ^ 1, j + 1);   // overlap next K/V tile DMA

    // S = Q K^T  (wave rows x 64 keys) -- preload all B frags, then batch WMMAs
    v8f sacc[4] = {};
#pragma unroll
    for (int ks = 0; ks < HD; ks += 32) {
      v16bf af = ldfragA(&Qs[wave * 16 + arow][ks], akb);
      v16bf bfr[4];
#pragma unroll
      for (int t = 0; t < 4; ++t)
        bfr[t] = ldfragB(&Ks[cur][t * 16 + (lane & 15)][ks], bkb);
#pragma unroll
      for (int t = 0; t < 4; ++t)
        sacc[t] = wmma_bf16(af, bfr[t], sacc[t]);
    }

    // online softmax update
#pragma unroll
    for (int e = 0; e < 8; ++e) {
      int qi = qt * 64 + wave * 16 + e + halfofs;
      float s[4];
      float mx = -1e30f;
#pragma unroll
      for (int t = 0; t < 4; ++t) {
        int ki = j * 64 + t * 16 + (lane & 15);
        float sv = sacc[t][e] * 0.125f;            // HD^-0.5
        if (ki > qi) sv = -1e30f;                  // causal mask
        s[t] = sv;
        mx = fmaxf(mx, sv);
      }
#pragma unroll
      for (int mk = 1; mk < 16; mk <<= 1) mx = fmaxf(mx, __shfl_xor(mx, mk, 16));
      float mnew  = fmaxf(mrow[e], mx);
      float scale = __expf(mrow[e] - mnew);
      float psum = 0.f;
#pragma unroll
      for (int t = 0; t < 4; ++t) { s[t] = __expf(s[t] - mnew); psum += s[t]; }
#pragma unroll
      for (int mk = 1; mk < 16; mk <<= 1) psum += __shfl_xor(psum, mk, 16);
      lrow[e] = lrow[e] * scale + psum;
      mrow[e] = mnew;
#pragma unroll
      for (int t = 0; t < 4; ++t) {
        oacc[t][e] = oacc[t][e] * scale;
        Ps[wave][e + halfofs][t * 16 + (lane & 15)] = (bf16)s[t];  // wave-private
      }
    }

    // O += P @ V  (A = P 16x64 over keys, B = V^T so per-column K is contiguous)
#pragma unroll
    for (int ks = 0; ks < 64; ks += 32) {
      v16bf af = ldfragA(&Ps[wave][arow][ks], akb);
      v16bf bfr[4];
#pragma unroll
      for (int t = 0; t < 4; ++t)
        bfr[t] = ldfragB(&Vs[cur][t * 16 + (lane & 15)][ks], bkb);
#pragma unroll
      for (int t = 0; t < 4; ++t)
        oacc[t] = wmma_bf16(af, bfr[t], oacc[t]);
    }
    async_join();
    __syncthreads();
  }

  // write normalized output
#pragma unroll
  for (int t = 0; t < 4; ++t) {
    int hd0 = t * 16 + (lane & 15);
#pragma unroll
    for (int e = 0; e < 8; ++e) {
      int qi = qt * 64 + wave * 16 + e + halfofs;
      O[base + (size_t)qi * C + hd0] = (bf16)(oacc[t][e] / lrow[e]);
    }
  }
}

// ---------------------------------------------------------------------------
// Host orchestration
// ---------------------------------------------------------------------------
extern "C" void kernel_launch(void* const* d_in, const int* in_sizes, int n_in,
                              void* d_out, int out_size, void* d_ws, size_t ws_size,
                              hipStream_t stream) {
  (void)in_sizes; (void)n_in; (void)out_size; (void)ws_size;

  const int*   idx   = (const int*)  d_in[0];
  const float* emb   = (const float*)d_in[1];
  const float* pos   = (const float*)d_in[2];
  const float* Wq    = (const float*)d_in[3];
  const float* Wk    = (const float*)d_in[4];
  const float* Wv    = (const float*)d_in[5];
  const float* Wo    = (const float*)d_in[6];
  const float* bo    = (const float*)d_in[7];
  const float* W1    = (const float*)d_in[8];
  const float* b1    = (const float*)d_in[9];
  const float* W2    = (const float*)d_in[10];
  const float* b2    = (const float*)d_in[11];
  const float* ln1g  = (const float*)d_in[12];
  const float* ln1b  = (const float*)d_in[13];
  const float* ln2g  = (const float*)d_in[14];
  const float* ln2b  = (const float*)d_in[15];
  const float* lnfg  = (const float*)d_in[16];
  const float* lnfb  = (const float*)d_in[17];
  const float* Wlm   = (const float*)d_in[18];
  const float* blm   = (const float*)d_in[19];
  float* logits = (float*)d_out;

  // workspace carve-up (256B aligned)
  char* wsp = (char*)d_ws;
  auto carve = [&](size_t bytes) {
    void* p = (void*)wsp;
    wsp += (bytes + 255) & ~(size_t)255;
    return p;
  };
  float* x  = (float*)carve((size_t)M * C * 4);
  bf16* h   = (bf16*)carve((size_t)M * C * 2);
  bf16* qb  = (bf16*)carve((size_t)M * C * 2);
  bf16* kb  = (bf16*)carve((size_t)M * C * 2);
  bf16* vb  = (bf16*)carve((size_t)M * C * 2);
  bf16* ao  = (bf16*)carve((size_t)M * C * 2);
  bf16* ff  = (bf16*)carve((size_t)M * FF * 2);
  bf16* wqb = (bf16*)carve((size_t)L * C * C * 2);
  bf16* wkb = (bf16*)carve((size_t)L * C * C * 2);
  bf16* wvb = (bf16*)carve((size_t)L * C * C * 2);
  bf16* wob = (bf16*)carve((size_t)L * C * C * 2);
  bf16* w1b = (bf16*)carve((size_t)L * C * FF * 2);
  bf16* w2b = (bf16*)carve((size_t)L * FF * C * 2);
  bf16* wlb = (bf16*)carve((size_t)C * Vv_ * 2);

  // weight conversion fp32 -> bf16
  dim3 cb(256), cg(1024);
  k_f32_to_bf16<<<cg, cb, 0, stream>>>(Wq,  wqb, (size_t)L * C * C);
  k_f32_to_bf16<<<cg, cb, 0, stream>>>(Wk,  wkb, (size_t)L * C * C);
  k_f32_to_bf16<<<cg, cb, 0, stream>>>(Wv,  wvb, (size_t)L * C * C);
  k_f32_to_bf16<<<cg, cb, 0, stream>>>(Wo,  wob, (size_t)L * C * C);
  k_f32_to_bf16<<<cg, cb, 0, stream>>>(W1,  w1b, (size_t)L * C * FF);
  k_f32_to_bf16<<<cg, cb, 0, stream>>>(W2,  w2b, (size_t)L * FF * C);
  k_f32_to_bf16<<<cg, cb, 0, stream>>>(Wlm, wlb, (size_t)C * Vv_);

  // embedding
  k_embed<<<M, 128, 0, stream>>>(idx, emb, pos, x);

  const dim3 lnGrid(M / 8), lnBlk(256);
  const dim3 gemmBlk(256);
  const dim3 gridC (C  / 64, M / 128);   // N=384
  const dim3 gridFF(FF / 64, M / 128);   // N=1536
  const dim3 gridV ((Vv_ + 63) / 64, M / 128);
  const dim3 attnGrid(T / 64, H, Bb), attnBlk(128);

  for (int l = 0; l < L; ++l) {
    // ---- attention sub-block ----
    k_layernorm<<<lnGrid, lnBlk, 0, stream>>>(x, ln1g + l * C, ln1b + l * C, h);
    k_gemm_bf16<0><<<gridC, gemmBlk, 0, stream>>>(h, wqb + (size_t)l * C * C, nullptr, qb, nullptr, C, C);
    k_gemm_bf16<0><<<gridC, gemmBlk, 0, stream>>>(h, wkb + (size_t)l * C * C, nullptr, kb, nullptr, C, C);
    k_gemm_bf16<0><<<gridC, gemmBlk, 0, stream>>>(h, wvb + (size_t)l * C * C, nullptr, vb, nullptr, C, C);
    k_attention<<<attnGrid, attnBlk, 0, stream>>>(qb, kb, vb, ao);
    k_gemm_bf16<2><<<gridC, gemmBlk, 0, stream>>>(ao, wob + (size_t)l * C * C, bo + l * C, nullptr, x, C, C);
    // ---- MLP sub-block ----
    k_layernorm<<<lnGrid, lnBlk, 0, stream>>>(x, ln2g + l * C, ln2b + l * C, h);
    k_gemm_bf16<1><<<gridFF, gemmBlk, 0, stream>>>(h, w1b + (size_t)l * C * FF, b1 + l * FF, ff, nullptr, FF, C);
    k_gemm_bf16<2><<<gridC, gemmBlk, 0, stream>>>(ff, w2b + (size_t)l * FF * C, b2 + l * C, nullptr, x, C, FF);
  }

  // final LN + LM head (fp32 logits to d_out)
  k_layernorm<<<lnGrid, lnBlk, 0, stream>>>(x, lnfg, lnfb, h);
  k_gemm_bf16<3><<<gridV, gemmBlk, 0, stream>>>(h, wlb, blm, nullptr, logits, Vv_, C);
}
